// ParagraphGAT_23965917512225
// MI455X (gfx1250) — matrix operand order repaired and checked by reading, not compiled, not run on hardware
//
#include <hip/hip_runtime.h>
#include <hip/hip_bf16.h>

// ---------------------------------------------------------------------------
// GATv2 x3 for MI455X (gfx1250, wave32). GEMMs via v_wmma_f32_16x16x32_bf16,
// edge phase via wave-per-(edge,head) reductions + fp32 atomics.
// ---------------------------------------------------------------------------

typedef __attribute__((ext_vector_type(16))) __bf16 v16bf;
typedef __attribute__((ext_vector_type(8)))  float  v8f;

#define DFEAT 128
#define NEG_SLOPE 0.2f

__device__ __forceinline__ unsigned short f2bf(float f) {
    unsigned u = __float_as_uint(f);
    unsigned rnd = 0x7fffu + ((u >> 16) & 1u);   // round-to-nearest-even
    return (unsigned short)((u + rnd) >> 16);
}

// monotonic float<->uint mapping so atomicMax(unsigned) == float max
__device__ __forceinline__ unsigned f2ord(float f) {
    unsigned u = __float_as_uint(f);
    return (u & 0x80000000u) ? ~u : (u | 0x80000000u);
}
__device__ __forceinline__ float ord2f(unsigned u) {
    return (u & 0x80000000u) ? __uint_as_float(u & 0x7fffffffu)
                             : __uint_as_float(~u);
}

// ---------------------------------------------------------------------------
// GEMM: Y[M x Nc] = X[M x 128] * W[128 x Nc], fp32 in/out, bf16 WMMA compute.
// Block = 128 threads (4 wave32). Block tile 32(M) x 64(N), full K=128 in LDS.
// Wave w: rows m0=(w&1)*16, cols n0=(w>>1)*32 (two 16x16 accumulators).
// ---------------------------------------------------------------------------
__global__ __launch_bounds__(128) void gemm_bf16_wmma(
    const float* __restrict__ X, const float* __restrict__ W,
    float* __restrict__ Y, int M, int Nc)
{
    __shared__ unsigned short As[32][132];   // A tile, row-major [m][k], padded
    __shared__ unsigned short Bs[64][132];   // B tile, [n][k] (transposed), padded

    const int tid = threadIdx.x;
    const int bm  = blockIdx.x * 32;
    const int bn  = blockIdx.y * 64;

    // stage A: 32x128, fp32 -> bf16, coalesced global reads
    for (int i = tid; i < 32 * 128; i += 128) {
        int r = i >> 7, c = i & 127;
        float v = (bm + r < M) ? X[(size_t)(bm + r) * DFEAT + c] : 0.0f;
        As[r][c] = f2bf(v);
    }
    // stage B: 128x64 read row-major from W, store [n][k]
    for (int i = tid; i < 128 * 64; i += 128) {
        int r = i >> 6, c = i & 63;                       // r = k, c = n
        Bs[c][r] = f2bf(W[(size_t)r * Nc + bn + c]);
    }
    __syncthreads();

    const int wave = tid >> 5;
    const int lane = tid & 31;
    const int lhi  = lane >> 4;          // 0: lanes 0-15, 1: lanes 16-31
    const int m0   = (wave & 1) * 16;
    const int n0   = (wave >> 1) * 32;
    const int mrow = m0 + (lane & 15);
    const int nloc0 = n0 + (lane & 15);
    const int nloc1 = nloc0 + 16;

    v8f acc0 = {}; v8f acc1 = {};

    union Frag { unsigned u32[8]; v16bf v; };

    #pragma unroll
    for (int k0 = 0; k0 < 128; k0 += 32) {
        Frag fa, fb0, fb1;
        #pragma unroll
        for (int j = 0; j < 8; ++j) {
            // A fragment: lanes<16 K=0..15, lanes>=16 K=16..31
            int ka = 2 * j + lhi * 16;
            fa.u32[j]  = *(const unsigned*)&As[mrow][k0 + ka];
            // B fragment: lanes<16 K=0..15, lanes>=16 K=16..31
            int kb = 2 * j + lhi * 16;
            fb0.u32[j] = *(const unsigned*)&Bs[nloc0][k0 + kb];
            fb1.u32[j] = *(const unsigned*)&Bs[nloc1][k0 + kb];
        }
        acc0 = __builtin_amdgcn_wmma_f32_16x16x32_bf16(
                   false, fa.v, false, fb0.v, (short)0, acc0, false, false);
        acc1 = __builtin_amdgcn_wmma_f32_16x16x32_bf16(
                   false, fa.v, false, fb1.v, (short)0, acc1, false, false);
    }

    // D layout: VGPR r -> row (r + lhi*8), col = lane&15
    const int nc = lane & 15;
    #pragma unroll
    for (int r = 0; r < 8; ++r) {
        int m = bm + m0 + r + lhi * 8;
        if (m < M) {
            size_t row = (size_t)m * Nc;
            Y[row + bn + n0 + nc]      = acc0[r];
            Y[row + bn + n0 + 16 + nc] = acc1[r];
        }
    }
}

// ---------------------------------------------------------------------------
// init: zero agg [N*128], emax (ordered-uint 0 == -inf-ish), denom [N*H]
// ---------------------------------------------------------------------------
__global__ void init_kernel(unsigned* emax, float* denom, float* agg,
                            int Nn, int H)
{
    int idx = blockIdx.x * blockDim.x + threadIdx.x;
    if (idx < Nn * DFEAT) agg[idx] = 0.0f;
    if (idx < Nn * H) { emax[idx] = 0u; denom[idx] = 0.0f; }
}

// ---------------------------------------------------------------------------
// score: one wave32 per (edge, head).
// e = sum_c leaky_relu(xl[src,h,c] + xr[dst,h,c]) * att[h,c]
// ---------------------------------------------------------------------------
__global__ __launch_bounds__(256) void edge_score_kernel(
    const float* __restrict__ xl, const float* __restrict__ xr,
    const float* __restrict__ att, const long long* __restrict__ ei,
    int E, int Nn, int H, float* __restrict__ escr, unsigned* __restrict__ emax)
{
    int wid  = (blockIdx.x * blockDim.x + threadIdx.x) >> 5;
    int lane = threadIdx.x & 31;
    int total = (E + Nn) * H;
    if (wid >= total) return;
    int e = wid / H, h = wid - e * H;
    int s, d;
    if (e < E) { s = (int)ei[e]; d = (int)ei[E + e]; } else { s = d = e - E; }

    const float* pl = xl + ((size_t)s * H + h) * DFEAT;
    const float* pr = xr + ((size_t)d * H + h) * DFEAT;
    const float* pa = att + (size_t)h * DFEAT;

    float sum = 0.0f;
    #pragma unroll
    for (int j = 0; j < 4; ++j) {
        int c = lane + 32 * j;
        float m = pl[c] + pr[c];
        m = (m > 0.0f) ? m : NEG_SLOPE * m;
        sum += m * pa[c];
    }
    #pragma unroll
    for (int off = 16; off > 0; off >>= 1)
        sum += __shfl_xor(sum, off, 32);

    if (lane == 0) {
        escr[(size_t)e * H + h] = sum;
        atomicMax(&emax[(size_t)d * H + h], f2ord(sum));
    }
}

// ---------------------------------------------------------------------------
// alpha: a = exp(e - emax[dst,h]); denom[dst,h] += a   (one thread per task)
// ---------------------------------------------------------------------------
__global__ void edge_alpha_kernel(const long long* __restrict__ ei,
                                  int E, int Nn, int H,
                                  const unsigned* __restrict__ emax,
                                  float* __restrict__ escr,
                                  float* __restrict__ denom)
{
    int idx = blockIdx.x * blockDim.x + threadIdx.x;
    int total = (E + Nn) * H;
    if (idx >= total) return;
    int e = idx / H, h = idx - e * H;
    int d = (e < E) ? (int)ei[E + e] : e - E;
    float mx = ord2f(emax[(size_t)d * H + h]);
    float a = __expf(escr[idx] - mx);
    escr[idx] = a;
    atomicAdd(&denom[(size_t)d * H + h], a);
}

// ---------------------------------------------------------------------------
// aggregate: block per edge, thread per channel; head-mean (1/H) folded in,
// so exactly one fp32 atomicAdd per (edge, channel).
// ---------------------------------------------------------------------------
__global__ __launch_bounds__(128) void edge_aggregate_kernel(
    const float* __restrict__ xl, const long long* __restrict__ ei,
    int E, int Nn, int H, const float* __restrict__ escr,
    const float* __restrict__ denom, float* __restrict__ agg)
{
    int e = blockIdx.x;
    int c = threadIdx.x;
    int s, d;
    if (e < E) { s = (int)ei[e]; d = (int)ei[E + e]; } else { s = d = e - E; }

    __shared__ float w[8];
    if (c < H) {
        float a  = escr[(size_t)e * H + c];
        float dn = denom[(size_t)d * H + c];
        w[c] = (a / dn) * (1.0f / (float)H);
    }
    __syncthreads();

    float sum = 0.0f;
    for (int h = 0; h < H; ++h)
        sum += w[h] * xl[((size_t)s * H + h) * DFEAT + c];

    atomicAdd(&agg[(size_t)d * DFEAT + c], sum);
}

// ---------------------------------------------------------------------------
// finalize: out = (elu?) elu(prev + agg + b) : prev + agg + b
// ---------------------------------------------------------------------------
__global__ void finalize_kernel(const float* __restrict__ prev,
                                const float* __restrict__ agg,
                                const float* __restrict__ b,
                                float* __restrict__ out, int Nn, int elu)
{
    int idx = blockIdx.x * blockDim.x + threadIdx.x;
    if (idx >= Nn * DFEAT) return;
    int c = idx & (DFEAT - 1);
    float v = prev[idx] + agg[idx] + b[c];
    if (elu) v = (v > 0.0f) ? v : (__expf(v) - 1.0f);
    out[idx] = v;
}

// ---------------------------------------------------------------------------
// host-side layer driver
// ---------------------------------------------------------------------------
static void run_layer(const float* hin, const float* Wl, const float* Wr,
                      const float* att, const float* b, int H, int elu,
                      const long long* ei, int E, int Nn,
                      float* xl, float* xr, float* escr, unsigned* emax,
                      float* denom, float* agg, float* hout,
                      hipStream_t stream)
{
    const int Nc = H * DFEAT;
    const int EE = E + Nn;

    {   // zero emax/denom/agg
        int n = Nn * DFEAT;
        init_kernel<<<(n + 255) / 256, 256, 0, stream>>>(emax, denom, agg, Nn, H);
    }
    {   // xl = hin @ Wl ; xr = hin @ Wr  (bf16 WMMA)
        dim3 g((Nn + 31) / 32, Nc / 64);
        gemm_bf16_wmma<<<g, 128, 0, stream>>>(hin, Wl, xl, Nn, Nc);
        gemm_bf16_wmma<<<g, 128, 0, stream>>>(hin, Wr, xr, Nn, Nc);
    }
    {   // attention scores + segment max
        long long waves = (long long)EE * H;
        long long thr = waves * 32;
        edge_score_kernel<<<(unsigned)((thr + 255) / 256), 256, 0, stream>>>(
            xl, xr, att, ei, E, Nn, H, escr, emax);
    }
    {   // softmax numerator + denominators
        long long t = (long long)EE * H;
        edge_alpha_kernel<<<(unsigned)((t + 255) / 256), 256, 0, stream>>>(
            ei, E, Nn, H, emax, escr, denom);
    }
    // weighted scatter-add (head mean folded in)
    edge_aggregate_kernel<<<EE, 128, 0, stream>>>(xl, ei, E, Nn, H,
                                                  escr, denom, agg);
    // residual + bias (+ ELU)
    finalize_kernel<<<(Nn * DFEAT + 255) / 256, 256, 0, stream>>>(
        hin, agg, b, hout, Nn, elu);
}

extern "C" void kernel_launch(void* const* d_in, const int* in_sizes, int n_in,
                              void* d_out, int out_size, void* d_ws, size_t ws_size,
                              hipStream_t stream)
{
    const float*     x   = (const float*)d_in[0];
    const long long* ei  = (const long long*)d_in[1];   // [2, E] int64
    const float* W1l = (const float*)d_in[2];
    const float* W1r = (const float*)d_in[3];
    const float* at1 = (const float*)d_in[4];
    const float* b1  = (const float*)d_in[5];
    const float* W2l = (const float*)d_in[6];
    const float* W2r = (const float*)d_in[7];
    const float* at2 = (const float*)d_in[8];
    const float* b2  = (const float*)d_in[9];
    const float* W3l = (const float*)d_in[10];
    const float* W3r = (const float*)d_in[11];
    const float* at3 = (const float*)d_in[12];
    const float* b3  = (const float*)d_in[13];

    const int Nn = in_sizes[0] / DFEAT;   // 50000
    const int E  = in_sizes[1] / 2;       // 100000
    const int EE = E + Nn;
    const int Hmax = 8;

    // workspace carve-out (~470 MB)
    char* w = (char*)d_ws;
    size_t off = 0;
    float*    xl    = (float*)(w + off); off += (size_t)Nn * Hmax * DFEAT * 4;
    float*    xr    = (float*)(w + off); off += (size_t)Nn * Hmax * DFEAT * 4;
    float*    escr  = (float*)(w + off); off += (size_t)EE * Hmax * 4;
    unsigned* emax  = (unsigned*)(w + off); off += (size_t)Nn * Hmax * 4;
    float*    denom = (float*)(w + off); off += (size_t)Nn * Hmax * 4;
    float*    agg   = (float*)(w + off); off += (size_t)Nn * DFEAT * 4;
    float*    hbuf  = (float*)(w + off); off += (size_t)Nn * DFEAT * 4;
    (void)ws_size; (void)n_in; (void)out_size;

    // layer 1: h = elu(x + gat1(x))
    run_layer(x,    W1l, W1r, at1, b1, 8, 1, ei, E, Nn,
              xl, xr, escr, emax, denom, agg, hbuf, stream);
    // layer 2: h = elu(h + gat2(h))   (in place on hbuf)
    run_layer(hbuf, W2l, W2r, at2, b2, 8, 1, ei, E, Nn,
              xl, xr, escr, emax, denom, agg, hbuf, stream);
    // layer 3: out = h + gat3(h)      (no elu)
    run_layer(hbuf, W3l, W3r, at3, b3, 4, 0, ei, E, Nn,
              xl, xr, escr, emax, denom, agg, (float*)d_out, stream);
}
